// StarTrans_69114613728828
// MI455X (gfx1250) — matrix operand order repaired and checked by reading, not compiled, not run on hardware
//
#include <hip/hip_runtime.h>

// ---------------------------------------------------------------------------
// Types for CDNA5 WMMA (wave32, v_wmma_f32_16x16x32_bf16)
// ---------------------------------------------------------------------------
typedef __bf16 bf16_t;
typedef bf16_t bf16x8  __attribute__((ext_vector_type(8)));
typedef bf16_t bf16x16 __attribute__((ext_vector_type(16)));
typedef float  f32x8   __attribute__((ext_vector_type(8)));

// ---------------------------------------------------------------------------
// WMMA GEMM: C[M,N] = A[M,K](bf16,row-major) * Wt[N,K](bf16,row-major)^T
//            (+bias[n]), epilogue modes:
//   mode 0 : C = acc + bias
//   mode 1 : C = addsrc + leaky_relu(acc + bias)      (slope 0.01)
// Block: 256 threads = 8 waves; waves tiled 2x4; wave tile = 32x32 (2x2 wmma)
// Block tile = 64 rows x 128 cols. Requires N % 128 == 0 (always 256 here).
// ---------------------------------------------------------------------------
__global__ __launch_bounds__(256)
void gemm_wmma_kernel(const bf16_t* __restrict__ A, int lda, int K,
                      const bf16_t* __restrict__ Wt,
                      const float* __restrict__ bias,
                      const float* __restrict__ addsrc,
                      float* __restrict__ C, int M, int N, int mode)
{
    const int lane = threadIdx.x & 31;
    const int wid  = threadIdx.x >> 5;
    const int half = lane >> 4;        // K-half select (ISA lane split)
    const int l16  = lane & 15;
    const int wm   = wid >> 2;         // 0..1
    const int wn   = wid & 3;          // 0..3
    const int row0 = blockIdx.y * 64 + wm * 32;
    const int col0 = blockIdx.x * 128 + wn * 32;

    // A row pointers (clamped so OOB lanes read valid memory; stores guarded)
    int m0 = row0 + l16;       if (m0 >= M) m0 = M - 1;
    int m1 = row0 + 16 + l16;  if (m1 >= M) m1 = M - 1;
    const bf16_t* ar0 = A + (size_t)m0 * lda + half * 16;
    const bf16_t* ar1 = A + (size_t)m1 * lda + half * 16;
    // B (pre-transposed weights): lane holds column n, contiguous K chunk
    const bf16_t* br0 = Wt + (size_t)(col0 + l16) * K + half * 16;
    const bf16_t* br1 = Wt + (size_t)(col0 + 16 + l16) * K + half * 16;

    f32x8 acc[2][2];
#pragma unroll
    for (int i = 0; i < 2; ++i)
#pragma unroll
        for (int j = 0; j < 2; ++j)
#pragma unroll
            for (int r = 0; r < 8; ++r) acc[i][j][r] = 0.0f;

    for (int k0 = 0; k0 < K; k0 += 32) {
        bf16x8 a0lo = *(const bf16x8*)(ar0 + k0);
        bf16x8 a0hi = *(const bf16x8*)(ar0 + k0 + 8);
        bf16x8 a1lo = *(const bf16x8*)(ar1 + k0);
        bf16x8 a1hi = *(const bf16x8*)(ar1 + k0 + 8);
        bf16x16 af0 = __builtin_shufflevector(a0lo, a0hi,
            0,1,2,3,4,5,6,7,8,9,10,11,12,13,14,15);
        bf16x16 af1 = __builtin_shufflevector(a1lo, a1hi,
            0,1,2,3,4,5,6,7,8,9,10,11,12,13,14,15);
        bf16x16 bf0 = *(const bf16x16*)(br0 + k0);
        bf16x16 bf1 = *(const bf16x16*)(br1 + k0);

        acc[0][0] = __builtin_amdgcn_wmma_f32_16x16x32_bf16(
            false, af0, false, bf0, (short)0, acc[0][0], false, false);
        acc[0][1] = __builtin_amdgcn_wmma_f32_16x16x32_bf16(
            false, af0, false, bf1, (short)0, acc[0][1], false, false);
        acc[1][0] = __builtin_amdgcn_wmma_f32_16x16x32_bf16(
            false, af1, false, bf0, (short)0, acc[1][0], false, false);
        acc[1][1] = __builtin_amdgcn_wmma_f32_16x16x32_bf16(
            false, af1, false, bf1, (short)0, acc[1][1], false, false);
    }

    // C/D layout: lane l -> n = l&15, VGPR r -> m = r + 8*(l>>4)
    const int nc0 = col0 + l16;
    const int nc1 = col0 + 16 + l16;
    const float bv0 = bias[nc0];
    const float bv1 = bias[nc1];
#pragma unroll
    for (int i = 0; i < 2; ++i) {
        const int rbase = row0 + i * 16 + half * 8;
#pragma unroll
        for (int j = 0; j < 2; ++j) {
            const int nn = (j == 0) ? nc0 : nc1;
            const float bb = (j == 0) ? bv0 : bv1;
#pragma unroll
            for (int r = 0; r < 8; ++r) {
                const int m = rbase + r;
                if (m < M) {
                    float v = acc[i][j][r] + bb;
                    const size_t idx = (size_t)m * N + nn;
                    if (mode == 1) {
                        float lv = v > 0.0f ? v : 0.01f * v;
                        C[idx] = addsrc[idx] + lv;
                    } else {
                        C[idx] = v;
                    }
                }
            }
        }
    }
}

// ---------------------------------------------------------------------------
// Weight transpose + bf16 convert: Wt[z][n][k] = W[z][k][n], square H x H
// ---------------------------------------------------------------------------
__global__ __launch_bounds__(256)
void convw_kernel(const float* __restrict__ W, bf16_t* __restrict__ Wt, int H)
{
    __shared__ float tile[16][17];
    const float* Ws = W + (size_t)blockIdx.z * H * H;
    bf16_t* Wd = Wt + (size_t)blockIdx.z * H * H;
    const int kb = blockIdx.x * 16, nb = blockIdx.y * 16;
    tile[threadIdx.y][threadIdx.x] =
        Ws[(size_t)(kb + threadIdx.y) * H + nb + threadIdx.x];
    __syncthreads();
    Wd[(size_t)(nb + threadIdx.y) * H + kb + threadIdx.x] =
        (bf16_t)tile[threadIdx.x][threadIdx.y];
}

// Embedding fc weight: W[300,256] -> Wt[256,320] bf16 (zero padded K)
__global__ __launch_bounds__(256)
void convembw_kernel(const float* __restrict__ W, bf16_t* __restrict__ Wt)
{
    int idx = blockIdx.x * 256 + threadIdx.x;
    if (idx >= 256 * 320) return;
    int k = idx % 320, n = idx / 320;
    Wt[idx] = (k < 300) ? (bf16_t)W[(size_t)k * 256 + n] : (bf16_t)0.0f;
}

// Gather embedding rows into padded bf16 A[M,320]
__global__ __launch_bounds__(256)
void gather_kernel(const int* __restrict__ data, const float* __restrict__ emb,
                   bf16_t* __restrict__ A, int M)
{
    int idx = blockIdx.x * 256 + threadIdx.x;
    if (idx >= M * 320) return;
    int k = idx % 320, m = idx / 320;
    float v = (k < 300) ? emb[(size_t)data[m] * 300 + k] : 0.0f;
    A[idx] = (bf16_t)v;
}

__global__ __launch_bounds__(256)
void addpos_kernel(float* __restrict__ nodes, const float* __restrict__ pos,
                   int total, int L)
{
    int idx = blockIdx.x * 256 + threadIdx.x;
    if (idx >= total) return;
    int h = idx & 255;
    int l = (idx >> 8) % L;
    nodes[idx] += pos[(size_t)l * 256 + h];
}

__global__ __launch_bounds__(256)
void relay_mean_kernel(const float* __restrict__ nodes,
                       float* __restrict__ relay, int B, int L)
{
    int idx = blockIdx.x * 256 + threadIdx.x;
    if (idx >= B * 256) return;
    int h = idx & 255, b = idx >> 8;
    float s = 0.0f;
    for (int l = 0; l < L; ++l) s += nodes[((size_t)b * L + l) * 256 + h];
    relay[idx] = s * (1.0f / (float)L);
}

// LayerNorm over H=256, one block per row, bf16 output
__global__ __launch_bounds__(256)
void layernorm_kernel(const float* __restrict__ x, const float* __restrict__ g,
                      const float* __restrict__ b, bf16_t* __restrict__ out)
{
    __shared__ float red[256];
    const int row = blockIdx.x, tid = threadIdx.x;
    float v = x[(size_t)row * 256 + tid];
    red[tid] = v; __syncthreads();
    for (int s = 128; s > 0; s >>= 1) {
        if (tid < s) red[tid] += red[tid + s];
        __syncthreads();
    }
    float mean = red[0] * (1.0f / 256.0f); __syncthreads();
    float d = v - mean;
    red[tid] = d * d; __syncthreads();
    for (int s = 128; s > 0; s >>= 1) {
        if (tid < s) red[tid] += red[tid + s];
        __syncthreads();
    }
    float var = red[0] * (1.0f / 256.0f);
    float y = g[tid] * d * rsqrtf(var + 1e-5f) + b[tid];
    out[(size_t)row * 256 + tid] = (bf16_t)y;
}

// Small fp32 GEMM: out[32,256] = X[32,256] @ W[256,256] + bias, opt leaky
__global__ __launch_bounds__(256)
void gemm32_kernel(const float* __restrict__ X, const float* __restrict__ W,
                   const float* __restrict__ bias, float* __restrict__ out,
                   int leaky)
{
    const int b = blockIdx.x, h = threadIdx.x;
    const float* xr = X + (size_t)b * 256;
    float acc = bias[h];
    for (int k = 0; k < 256; ++k) acc += xr[k] * W[(size_t)k * 256 + h];
    if (leaky) acc = acc > 0.0f ? acc : 0.01f * acc;
    out[(size_t)b * 256 + h] = acc;
}

// msa1 ring attention (window 3 + relay key; VALUES from k, faithful to bug)
__global__ __launch_bounds__(256)
void attn1_kernel(const float* __restrict__ qbuf, const float* __restrict__ kbuf,
                  const float* __restrict__ akb, const float* __restrict__ avb,
                  bf16_t* __restrict__ att, int B, int L)
{
    int idx = blockIdx.x * 256 + threadIdx.x;
    if (idx >= B * L * 8) return;
    const int n  = idx & 7;
    const int bl = idx >> 3;
    const int l  = bl % L;
    const int b  = bl / L;

    float qv[32];
    {
        const float4* p = (const float4*)(qbuf + (size_t)bl * 256 + n * 32);
#pragma unroll
        for (int i = 0; i < 8; ++i) {
            float4 t = p[i];
            qv[4*i+0]=t.x; qv[4*i+1]=t.y; qv[4*i+2]=t.z; qv[4*i+3]=t.w;
        }
    }
    const float* kp[4];
    kp[0] = akb + (size_t)b * 256 + n * 32;
    kp[1] = (l > 0)     ? kbuf + ((size_t)bl - 1) * 256 + n * 32 : nullptr;
    kp[2] = kbuf + (size_t)bl * 256 + n * 32;
    kp[3] = (l + 1 < L) ? kbuf + ((size_t)bl + 1) * 256 + n * 32 : nullptr;

    const float scale = 0.17677669529663687f;  // 1/sqrt(32)
    float sc[4];
#pragma unroll
    for (int j = 0; j < 4; ++j) {
        float d = 0.0f;
        if (kp[j]) {
            const float4* p = (const float4*)kp[j];
#pragma unroll
            for (int i = 0; i < 8; ++i) {
                float4 t = p[i];
                d += qv[4*i+0]*t.x + qv[4*i+1]*t.y + qv[4*i+2]*t.z + qv[4*i+3]*t.w;
            }
        }
        sc[j] = d * scale;  // zero-padded keys -> score 0, in softmax (faithful)
    }
    float mx = fmaxf(fmaxf(sc[0], sc[1]), fmaxf(sc[2], sc[3]));
    float e0 = __expf(sc[0]-mx), e1 = __expf(sc[1]-mx);
    float e2 = __expf(sc[2]-mx), e3 = __expf(sc[3]-mx);
    float inv = 1.0f / (e0 + e1 + e2 + e3);
    float al[4] = {e0*inv, e1*inv, e2*inv, e3*inv};

    float out[32];
    {
        const float4* p = (const float4*)(avb + (size_t)b * 256 + n * 32);
#pragma unroll
        for (int i = 0; i < 8; ++i) {
            float4 t = p[i];
            out[4*i+0]=al[0]*t.x; out[4*i+1]=al[0]*t.y;
            out[4*i+2]=al[0]*t.z; out[4*i+3]=al[0]*t.w;
        }
    }
#pragma unroll
    for (int j = 1; j < 4; ++j) {
        if (!kp[j]) continue;   // zero value rows (pad) contribute nothing
        const float4* p = (const float4*)kp[j];
#pragma unroll
        for (int i = 0; i < 8; ++i) {
            float4 t = p[i];
            out[4*i+0]+=al[j]*t.x; out[4*i+1]+=al[j]*t.y;
            out[4*i+2]+=al[j]*t.z; out[4*i+3]+=al[j]*t.w;
        }
    }
    bf16_t* op = att + (size_t)bl * 256 + n * 32;
#pragma unroll
    for (int i = 0; i < 32; ++i) op[i] = (bf16_t)out[i];
}

// Build y = [relay ; nodes] as bf16, [B, L+1, 256]
__global__ __launch_bounds__(256)
void build_y_kernel(const float* __restrict__ relay,
                    const float* __restrict__ nodes,
                    bf16_t* __restrict__ y, int B, int L)
{
    int idx = blockIdx.x * 256 + threadIdx.x;
    const int L1 = L + 1;
    if (idx >= B * L1 * 256) return;
    int h = idx & 255;
    int r = idx >> 8;
    int b = r / L1, l = r % L1;
    float v = (l == 0) ? relay[(size_t)b * 256 + h]
                       : nodes[((size_t)b * L + (l - 1)) * 256 + h];
    y[idx] = (bf16_t)v;
}

// msa2: relay attends over 513 keys. One block per (b, head).
__global__ __launch_bounds__(256)
void attn2_kernel(const float* __restrict__ q2, const float* __restrict__ k2,
                  const float* __restrict__ v2, float* __restrict__ att2,
                  int L1)
{
    __shared__ float sc[513];
    __shared__ float red[256];
    __shared__ float part[8][32];
    __shared__ float qs[32];
    const int b = blockIdx.x >> 3;
    const int n = blockIdx.x & 7;
    const int tid = threadIdx.x;
    if (tid < 32) qs[tid] = q2[(size_t)b * 256 + n * 32 + tid];
    __syncthreads();

    const float scale = 0.17677669529663687f;
    float lmax = -3.4e38f;
    for (int j = tid; j < L1; j += 256) {
        const float* kp = k2 + ((size_t)b * L1 + j) * 256 + n * 32;
        float d = 0.0f;
#pragma unroll
        for (int c = 0; c < 32; ++c) d += qs[c] * kp[c];
        d *= scale;
        sc[j] = d;
        lmax = fmaxf(lmax, d);
    }
    red[tid] = lmax; __syncthreads();
    for (int s = 128; s > 0; s >>= 1) {
        if (tid < s) red[tid] = fmaxf(red[tid], red[tid + s]);
        __syncthreads();
    }
    const float mx = red[0]; __syncthreads();

    float lsum = 0.0f;
    for (int j = tid; j < L1; j += 256) {
        float e = __expf(sc[j] - mx);
        sc[j] = e;
        lsum += e;
    }
    red[tid] = lsum; __syncthreads();
    for (int s = 128; s > 0; s >>= 1) {
        if (tid < s) red[tid] += red[tid + s];
        __syncthreads();
    }
    const float inv = 1.0f / red[0]; __syncthreads();

    const int g = tid >> 5, d = tid & 31;
    float acc = 0.0f;
    for (int j = g; j < L1; j += 8)
        acc += sc[j] * v2[((size_t)b * L1 + j) * 256 + n * 32 + d];
    part[g][d] = acc; __syncthreads();
    if (tid < 32) {
        float s = 0.0f;
#pragma unroll
        for (int gg = 0; gg < 8; ++gg) s += part[gg][tid];
        att2[(size_t)b * 256 + n * 32 + tid] = s * inv;
    }
}

__global__ __launch_bounds__(256)
void mask_kernel(const int* __restrict__ data, float* __restrict__ nodes,
                 int total)
{
    int idx = blockIdx.x * 256 + threadIdx.x;
    if (idx >= total) return;
    if (data[idx >> 8] == 1) nodes[idx] = 0.0f;
}

__global__ __launch_bounds__(256)
void final_kernel(const float* __restrict__ nodes, const float* __restrict__ relay,
                  float* __restrict__ out, int B, int L)
{
    int idx = blockIdx.x * 256 + threadIdx.x;
    if (idx >= B * 256) return;
    int h = idx & 255, b = idx >> 8;
    float m = -3.4e38f;
    for (int l = 0; l < L; ++l)
        m = fmaxf(m, nodes[((size_t)b * L + l) * 256 + h]);
    out[idx] = 0.5f * m + 0.5f * relay[idx];
}

// ---------------------------------------------------------------------------
// Host driver
// ---------------------------------------------------------------------------
extern "C" void kernel_launch(void* const* d_in, const int* in_sizes, int n_in,
                              void* d_out, int out_size, void* d_ws, size_t ws_size,
                              hipStream_t stream)
{
    (void)in_sizes; (void)n_in; (void)out_size; (void)ws_size;
    const int B = 32, L = 512, H = 256, ITERS = 6;
    const int M  = B * L;        // 16384
    const int L1 = L + 1;        // 513
    const int M2 = B * L1;       // 16416
    const int EP = 320;          // padded embedding K

    const int*   data  = (const int*)d_in[0];
    const float* emb   = (const float*)d_in[1];
    const float* embW  = (const float*)d_in[2];
    const float* embB  = (const float*)d_in[3];
    const float* pos   = (const float*)d_in[4];
    const float* ln_g  = (const float*)d_in[5];
    const float* ln_b  = (const float*)d_in[6];
    const float* r_WQ  = (const float*)d_in[7];  const float* r_bQ = (const float*)d_in[8];
    const float* r_WK  = (const float*)d_in[9];  const float* r_bK = (const float*)d_in[10];
    const float* r_WV  = (const float*)d_in[11]; const float* r_bV = (const float*)d_in[12];
    const float* r_WO  = (const float*)d_in[13]; const float* r_bO = (const float*)d_in[14];
    const float* s_WQ  = (const float*)d_in[15]; const float* s_bQ = (const float*)d_in[16];
    const float* s_WK  = (const float*)d_in[17]; const float* s_bK = (const float*)d_in[18];
    const float* s_WV  = (const float*)d_in[19]; const float* s_bV = (const float*)d_in[20];
    const float* s_WO  = (const float*)d_in[21]; const float* s_bO = (const float*)d_in[22];

    char* ws = (char*)d_ws;
    size_t off = 0;
    auto take = [&](size_t bytes) -> char* {
        char* p = ws + off;
        off += (bytes + 255) & ~(size_t)255;
        return p;
    };

    const size_t WB = (size_t)ITERS * H * H * sizeof(bf16_t);
    bf16_t* wt_rQ  = (bf16_t*)take(WB);
    bf16_t* wt_rK  = (bf16_t*)take(WB);
    bf16_t* wt_rO  = (bf16_t*)take(WB);
    bf16_t* wt_sK  = (bf16_t*)take(WB);
    bf16_t* wt_sV  = (bf16_t*)take(WB);
    bf16_t* wt_emb = (bf16_t*)take((size_t)H * EP * sizeof(bf16_t));

    float*  nodes = (float*) take((size_t)M * H * sizeof(float));
    bf16_t* nln   = (bf16_t*)take((size_t)M * H * sizeof(bf16_t));
    // Aliased buffers: q aliases k2 (and gathered embedding A), k aliases v2
    char*   bufA  = take((size_t)M2 * H * sizeof(float));   // 16.8 MB
    char*   bufBc = take((size_t)M2 * H * sizeof(float));   // 16.8 MB
    float*  qb  = (float*)bufA;   float* k2 = (float*)bufA;  bf16_t* Aemb = (bf16_t*)bufA;
    float*  kb  = (float*)bufBc;  float* v2 = (float*)bufBc;
    bf16_t* att1 = (bf16_t*)take((size_t)M * H * sizeof(bf16_t));
    bf16_t* ybuf = (bf16_t*)take((size_t)M2 * H * sizeof(bf16_t));
    float*  relay = (float*)take((size_t)B * H * sizeof(float));
    float*  akb   = (float*)take((size_t)B * H * sizeof(float));
    float*  avb   = (float*)take((size_t)B * H * sizeof(float));
    float*  q2b   = (float*)take((size_t)B * H * sizeof(float));
    float*  att2b = (float*)take((size_t)B * H * sizeof(float));

    const dim3 blk256(256);
    const dim3 gemm_grid(2, (M + 63) / 64);     // N=256, M=16384
    const dim3 gemm_grid2(2, (M2 + 63) / 64);   // M=16416

    // ---- weight conversion (bf16, transposed) ----
    {
        dim3 g(H / 16, H / 16, ITERS), b(16, 16);
        convw_kernel<<<g, b, 0, stream>>>(r_WQ, wt_rQ, H);
        convw_kernel<<<g, b, 0, stream>>>(r_WK, wt_rK, H);
        convw_kernel<<<g, b, 0, stream>>>(r_WO, wt_rO, H);
        convw_kernel<<<g, b, 0, stream>>>(s_WK, wt_sK, H);
        convw_kernel<<<g, b, 0, stream>>>(s_WV, wt_sV, H);
    }
    convembw_kernel<<<(256 * EP + 255) / 256, blk256, 0, stream>>>(embW, wt_emb);

    // ---- embedding + fc + pos ----
    gather_kernel<<<((size_t)M * EP + 255) / 256, blk256, 0, stream>>>(data, emb, Aemb, M);
    gemm_wmma_kernel<<<gemm_grid, blk256, 0, stream>>>(
        Aemb, EP, EP, wt_emb, embB, nullptr, nodes, M, H, 0);
    addpos_kernel<<<(M * H + 255) / 256, blk256, 0, stream>>>(nodes, pos, M * H, L);
    relay_mean_kernel<<<(B * H + 255) / 256, blk256, 0, stream>>>(nodes, relay, B, L);

    for (int i = 0; i < ITERS; ++i) {
        const size_t wOff = (size_t)i * H * H;
        const size_t bOff = (size_t)i * H;

        // n_ln = LayerNorm(nodes)  (bf16 out)
        layernorm_kernel<<<M, blk256, 0, stream>>>(nodes, ln_g + bOff, ln_b + bOff, nln);

        // q = n_ln @ r_WQ + b ; k = n_ln @ r_WK + b       (WMMA bf16)
        gemm_wmma_kernel<<<gemm_grid, blk256, 0, stream>>>(
            nln, H, H, wt_rQ + wOff, r_bQ + bOff, nullptr, qb, M, H, 0);
        gemm_wmma_kernel<<<gemm_grid, blk256, 0, stream>>>(
            nln, H, H, wt_rK + wOff, r_bK + bOff, nullptr, kb, M, H, 0);

        // relay key/value projections (tiny fp32 GEMMs)
        gemm32_kernel<<<B, blk256, 0, stream>>>(relay, r_WK + wOff, r_bK + bOff, akb, 0);
        gemm32_kernel<<<B, blk256, 0, stream>>>(relay, r_WV + wOff, r_bV + bOff, avb, 0);

        // ring attention
        attn1_kernel<<<(B * L * 8 + 255) / 256, blk256, 0, stream>>>(
            qb, kb, akb, avb, att1, B, L);

        // nodes = nodes + leaky_relu(att @ r_WO + b)      (fused WMMA epilogue)
        gemm_wmma_kernel<<<gemm_grid, blk256, 0, stream>>>(
            att1, H, H, wt_rO + wOff, r_bO + bOff, nodes, nodes, M, H, 1);

        // y = [relay; nodes] (bf16)
        build_y_kernel<<<(B * L1 * H + 255) / 256, blk256, 0, stream>>>(
            relay, nodes, ybuf, B, L);

        // k2 = y @ s_WK + b ; v2 = y @ s_WV + b           (WMMA, M=16416)
        gemm_wmma_kernel<<<gemm_grid2, blk256, 0, stream>>>(
            ybuf, H, H, wt_sK + wOff, s_bK + bOff, nullptr, k2, M2, H, 0);
        gemm_wmma_kernel<<<gemm_grid2, blk256, 0, stream>>>(
            ybuf, H, H, wt_sV + wOff, s_bV + bOff, nullptr, v2, M2, H, 0);

        // q2 = relay @ s_WQ + b  (tiny)
        gemm32_kernel<<<B, blk256, 0, stream>>>(relay, s_WQ + wOff, s_bQ + bOff, q2b, 0);

        // relay attention over 513 keys
        attn2_kernel<<<B * 8, blk256, 0, stream>>>(q2b, k2, v2, att2b, L1);

        // relay = leaky_relu(att2 @ s_WO + b)  (tiny)
        gemm32_kernel<<<B, blk256, 0, stream>>>(att2b, s_WO + wOff, s_bO + bOff, relay, 1);

        // zero masked nodes
        mask_kernel<<<(M * H + 255) / 256, blk256, 0, stream>>>(data, nodes, M * H);
    }

    // rep = 0.5 * max_l(nodes) + 0.5 * relay
    final_kernel<<<(B * H + 255) / 256, blk256, 0, stream>>>(
        nodes, relay, (float*)d_out, B, L);
}
